// CARAFEUpsampler_86466281603203
// MI455X (gfx1250) — matrix operand order repaired and checked by reading, not compiled, not run on hardware
//
#include <hip/hip_runtime.h>
#include <hip/hip_bf16.h>

// CARAFE upsampler forward for gfx1250 (MI455X), wave32 + WMMA.
// Shapes (from reference): N=2, C=256, H=W=64, r=2, UP_K=5, ENC_K=3,
// comp=64, enc_out=100. Output (2,256,128,128) fp32.

#define NB   2
#define CIN  256
#define HH   64
#define WW   64
#define HW   (HH * WW)          // 4096
#define COMP 64
#define ENC  100
#define ENCP 112                // padded to 7x16
#define KENC (9 * COMP)         // 576
#define SH   128
#define SW   128
#define SHW  (SH * SW)          // 16384

typedef __attribute__((ext_vector_type(16))) _Float16 v16h;
typedef __attribute__((ext_vector_type(8)))  float    v8f;

__device__ __forceinline__ v8f wmma_f16(v16h a, v16h b, v8f c) {
  // (neg_a, A, neg_b, B, c_mod, C, reuse_a, reuse_b) -> v_wmma_f32_16x16x32_f16
  return __builtin_amdgcn_wmma_f32_16x16x32_f16(false, a, false, b,
                                                (short)0, c, false, false);
}

// ---------------------------------------------------------------------------
// Kernel 1: compressed[n][p][co] = b[co] + sum_ci w_comp[co][ci]*feat[n][ci][p]
// GEMM: M = 16 spatial positions / block, N = 64 out channels (4 waves x 16),
//       K = 256 in 8 WMMA steps of 32.
// grid = (NB*HW)/16 = 512 blocks, block = 128 threads (4 waves).
// ---------------------------------------------------------------------------
__global__ __launch_bounds__(128)
void carafe_compress(const float* __restrict__ feat,
                     const float* __restrict__ w_comp,
                     const float* __restrict__ b_comp,
                     float* __restrict__ comp_ws) {
  const int lane = threadIdx.x & 31;
  const int wave = threadIdx.x >> 5;       // 0..3 -> co tile
  const int l    = lane & 15;
  const int hi   = lane >> 4;              // lane half

  const int g0 = blockIdx.x * 16;          // global spatial tile start
  const int n  = g0 >> 12;                 // 4096 positions per batch
  const int p0 = g0 & (HW - 1);
  const int co0 = wave * 16;

  const float* fsrc = feat + (size_t)n * CIN * HW + p0 + l;      // A: M = p0+l
  const float* wsrc = w_comp + (size_t)(co0 + l) * CIN;          // B: N = co0+l

  v8f acc = {};
  for (int k0 = 0; k0 < CIN; k0 += 32) {
    v16h a, b;
#pragma unroll
    for (int e = 0; e < 16; ++e) {
      const int ka = e + hi * 16;                      // A-frag K map (16-bit)
      a[e] = (_Float16)fsrc[(size_t)(k0 + ka) * HW];
    }
#pragma unroll
    for (int e = 0; e < 16; ++e) {
      const int kb = e + hi * 16;                      // B-frag K map (16-bit)
      b[e] = (_Float16)wsrc[k0 + kb];
    }
    acc = wmma_f16(a, b, acc);
  }

  const float bias = b_comp[co0 + l];
  float* dst = comp_ws + (size_t)n * HW * COMP + co0 + l;
#pragma unroll
  for (int r = 0; r < 8; ++r) {
    const int p = p0 + r + 8 * hi;                     // C/D: M = r + 8*half
    dst[(size_t)p * COMP] = acc[r] + bias;
  }
}

// ---------------------------------------------------------------------------
// Kernel 2: kp[n][p][e] = sum_{ci,dy,dx} w_enc[e][ci][dy][dx] *
//                         compressed[n][(h+dy)(w+dx)][ci]    (pad=1)
// GEMM: M = 16 spatial / block, N = 112 (7 waves x 16, e>=100 zeroed),
//       K = 9*64 = 576 ordered (tap*64 + ci) so each 32-step has one tap.
// grid = 512 blocks, block = 224 threads (7 waves).
// ---------------------------------------------------------------------------
__global__ __launch_bounds__(224)
void carafe_encode(const float* __restrict__ comp_ws,
                   const float* __restrict__ w_enc,
                   float* __restrict__ kp_ws) {
  const int lane = threadIdx.x & 31;
  const int wave = threadIdx.x >> 5;       // 0..6 -> e tile
  const int l    = lane & 15;
  const int hi   = lane >> 4;

  const int g0 = blockIdx.x * 16;
  const int n  = g0 >> 12;
  const int p0 = g0 & (HW - 1);
  const int e0 = wave * 16;

  const int p = p0 + l;                    // this lane's M row (spatial)
  const int h = p >> 6;
  const int w = p & 63;

  const float* cbase = comp_ws + (size_t)n * HW * COMP;
  const int    e     = e0 + l;
  const bool   evld  = (e < ENC);
  const float* wsrc  = w_enc + (size_t)e * KENC;   // [e][ci][3][3]

  v8f acc = {};
  for (int k0 = 0; k0 < KENC; k0 += 32) {
    const int tap = k0 >> 6;               // 0..8, fixed within a 32-step
    const int ci0 = k0 & 63;               // 0 or 32
    const int dy  = tap / 3 - 1;
    const int dx  = tap % 3 - 1;
    const int hh  = h + dy;
    const int ww  = w + dx;
    const bool inb = (hh >= 0) & (hh < HH) & (ww >= 0) & (ww < WW);
    const float* asrc = cbase + (size_t)(hh * WW + ww) * COMP + ci0;

    v16h a, b;
#pragma unroll
    for (int ei = 0; ei < 16; ++ei) {
      const int ka = ei + hi * 16;
      a[ei] = (_Float16)(inb ? asrc[ka] : 0.0f);
    }
#pragma unroll
    for (int ei = 0; ei < 16; ++ei) {
      const int ci = ci0 + ei + hi * 16;
      b[ei] = (_Float16)(evld ? wsrc[ci * 9 + tap] : 0.0f);
    }
    acc = wmma_f16(a, b, acc);
  }

  float* dst = kp_ws + (size_t)n * HW * ENCP + e;
#pragma unroll
  for (int r = 0; r < 8; ++r) {
    const int pr = p0 + r + 8 * hi;
    dst[(size_t)pr * ENCP] = acc[r];
  }
}

// ---------------------------------------------------------------------------
// Kernel 3: pixel-shuffle + softmax(25) + CARAFE combine (all fp32).
// out[n,c,y,x] = sum_{i,j} feat[n,c,y/2+i-2,x/2+j-2] * softmax_k[ij]
// One block per (n, y, half-row of 64 x). 64 threads do softmax into LDS,
// then 256 threads = 64 px * 4 channel groups stream 64 channels each.
// grid = NB*SH*2 = 512 blocks, block = 256 threads.
// ---------------------------------------------------------------------------
__global__ __launch_bounds__(256)
void carafe_combine(const float* __restrict__ feat,
                    const float* __restrict__ kp_ws,
                    float* __restrict__ out) {
  __shared__ float wsm[64][26];            // 25 weights per pixel (+pad)

  const int n   = blockIdx.x >> 8;         // 256 blocks per batch
  const int rem = blockIdx.x & 255;
  const int y   = rem >> 1;
  const int x0  = (rem & 1) << 6;
  const int t   = threadIdx.x;

  if (t < 64) {
    const int x  = x0 + t;
    const int p  = ((y >> 1) << 6) + (x >> 1);
    const int eb = ((y & 1) << 1) + (x & 1);           // pixel-shuffle slot
    const float* src = kp_ws + (size_t)n * HW * ENCP + (size_t)p * ENCP + eb;
    float v[25];
    float m = -1e30f;
#pragma unroll
    for (int c = 0; c < 25; ++c) { v[c] = src[c * 4]; m = fmaxf(m, v[c]); }
    float s = 0.0f;
#pragma unroll
    for (int c = 0; c < 25; ++c) { v[c] = __expf(v[c] - m); s += v[c]; }
    const float inv = 1.0f / s;
#pragma unroll
    for (int c = 0; c < 25; ++c) wsm[t][c] = v[c] * inv;
  }
  __syncthreads();

  const int px = t & 63;
  const int g  = t >> 6;                   // channel group 0..3
  const int x  = x0 + px;
  const int h0 = y >> 1;
  const int w0 = x >> 1;

  float wk[25];
#pragma unroll
  for (int c = 0; c < 25; ++c) wk[c] = wsm[px][c];

  const float* fb = feat + (size_t)n * CIN * HW;
  float* ob = out + (size_t)n * CIN * SHW + (size_t)y * SW + x;

  for (int cc = 0; cc < 64; ++cc) {
    const int c = g * 64 + cc;
    const float* fc = fb + (size_t)c * HW;
    float acc = 0.0f;
#pragma unroll
    for (int i = 0; i < 5; ++i) {
      const int hh = h0 + i - 2;
      if (hh < 0 || hh >= HH) continue;
#pragma unroll
      for (int j = 0; j < 5; ++j) {
        const int ww = w0 + j - 2;
        if (ww < 0 || ww >= WW) continue;
        acc = fmaf(fc[hh * WW + ww], wk[i * 5 + j], acc);
      }
    }
    ob[(size_t)c * SHW] = acc;
  }
}

// ---------------------------------------------------------------------------
extern "C" void kernel_launch(void* const* d_in, const int* in_sizes, int n_in,
                              void* d_out, int out_size, void* d_ws, size_t ws_size,
                              hipStream_t stream) {
  const float* feat   = (const float*)d_in[0];   // (2,256,64,64)
  const float* w_comp = (const float*)d_in[1];   // (64,256,1,1)
  const float* b_comp = (const float*)d_in[2];   // (64,)
  const float* w_enc  = (const float*)d_in[3];   // (100,64,3,3)
  float* out = (float*)d_out;                    // (2,256,128,128)

  float* comp_ws = (float*)d_ws;                       // NB*HW*COMP  = 512K f32
  float* kp_ws   = comp_ws + (size_t)NB * HW * COMP;   // NB*HW*ENCP  = 896K f32

  carafe_compress<<<(NB * HW) / 16, 128, 0, stream>>>(feat, w_comp, b_comp, comp_ws);
  carafe_encode  <<<(NB * HW) / 16, 224, 0, stream>>>(comp_ws, w_enc, kp_ws);
  carafe_combine <<<NB * SH * 2,    256, 0, stream>>>(feat, kp_ws, out);
}
